// MLDecoderClassificationHead_41901700939798
// MI455X (gfx1250) — matrix operand order repaired and checked by reading, not compiled, not run on hardware
//
#include <hip/hip_runtime.h>
#include <hip/hip_bf16.h>

// ---------------------------------------------------------------------------
// ML-Decoder classification head for MI455X (gfx1250, wave32, WMMA).
// - All heavy math: v_wmma_f32_16x16x32_bf16 (bf16 in, f32 accum).
// - Tile feed: GLOBAL_LOAD_ASYNC_TO_LDS_B128 (ASYNCcnt) with double-buffered
//   LDS so the next K-tile streams in while WMMAs run on the current one.
// - Weights converted to bf16 once per launch; activations stay bf16.
// - tgt/q are batch-invariant -> computed once (100 rows, not 6400).
// ---------------------------------------------------------------------------

#define DEV __device__ __forceinline__

typedef __attribute__((ext_vector_type(16))) __bf16 v16bf;
typedef __attribute__((ext_vector_type(8)))  float  v8f;

namespace cfg {
constexpr int Bn  = 64;    // batch
constexpr int CIN = 2048;
constexpr int S   = 196;   // 14*14
constexpr int D   = 768;
constexpr int FF  = 2048;
constexpr int G   = 100;
constexpr int NC  = 9605;
constexpr int NH  = 8;
constexpr int HD  = 96;
constexpr int DF  = 97;
constexpr int BS  = Bn * S;   // 12544
constexpr int BG  = Bn * G;   // 6400
}

DEV unsigned short f2bf(float f) {
  unsigned u = __builtin_bit_cast(unsigned, f);
  unsigned r = u + 0x7FFFu + ((u >> 16) & 1u);   // round-to-nearest-even
  return (unsigned short)(r >> 16);
}

// ---------------------------------------------------------------------------
// WMMA fragments (CDNA5 ISA 7.12.2 VGPR layouts, 16-bit operands, wave32)
// ---------------------------------------------------------------------------
union Frag { v16bf v; unsigned u[8]; };

constexpr int KT     = 32;        // K per wmma step (bf16)
// Row pitch 40 shorts = 80B: 16B-aligned rows (b128 async stores legal) and
// conflict-free fragment gathers (20*m mod 64 distinct for m=0..15).
constexpr int LDS_LD = 40;

// A 16x32: lane L -> row M=L&15; VGPR v -> K pair per ISA table
DEV Frag load_afrag(const unsigned short* base, int lane) {
  const unsigned short* row = base + (lane & 15) * LDS_LD;
  const int half = lane >> 4;
  Frag f;
#pragma unroll
  for (int v = 0; v < 8; ++v) {
    int k0 = ((v >> 2) << 4) + (half << 3) + ((v & 3) << 1);
    f.u[v] = *(const unsigned*)(row + k0);
  }
  return f;
}

// B 32x16: lane L -> col N=L&15; VGPR v -> K pair per ISA table (same as A)
DEV Frag load_bfrag(const unsigned short* base, int lane) {
  const unsigned short* row = base + (lane & 15) * LDS_LD;
  const int half = lane >> 4;
  Frag f;
#pragma unroll
  for (int v = 0; v < 8; ++v)
    f.u[v] = *(const unsigned*)(row + ((v >> 2) << 4) + (half << 3) + ((v & 3) << 1));
  return f;
}

DEV void wait_async0() {
#if __has_builtin(__builtin_amdgcn_s_wait_asynccnt)
  __builtin_amdgcn_s_wait_asynccnt(0);
#else
  asm volatile("s_wait_asynccnt 0x0" ::: "memory");
#endif
}

// Async full-tile loader: 128 rows x 32 bf16 via GLOBAL_LOAD_ASYNC_TO_LDS_B128
// (GVS mode: SGPR64 base + per-lane 32-bit byte offset). src includes the kb
// column offset; rs = row stride in elements (must be 16B-aligned rows).
DEV void load_tile_async(unsigned short* dst, const unsigned short* src,
                         long long rs, int row0, int tid) {
  const unsigned lbase = (unsigned)(size_t)dst;                 // LDS offset
  const unsigned long long gbase = (unsigned long long)(size_t)src;
#pragma unroll
  for (int i = 0; i < 2; ++i) {
    int idx = tid + i * 256;
    int r  = idx >> 2;
    int c8 = (idx & 3) * 8;                                     // shorts
    unsigned goff = (unsigned)(((long long)(row0 + r) * rs + c8) * 2);
    unsigned loff = lbase + (unsigned)(r * LDS_LD + c8) * 2;
    asm volatile("global_load_async_to_lds_b128 %0, %1, %2"
                 :: "v"(loff), "v"(goff), "s"(gbase) : "memory");
  }
}

// Fallback loader (edges / strided / f32 sources); zero-fills out of range.
template<int ROWS>
DEV void load_tile(unsigned short* dst, const void* src, int srcF32,
                   long long rs, long long cs, int row0, int rowMax,
                   int k0, int kMax, int tid) {
  if (!srcF32 && cs == 1) {              // contiguous-K: dword pair loads
    constexpr int DW = KT / 2;           // 16 dwords per row
    for (int d = tid; d < ROWS * DW; d += 256) {
      int r = d / DW, c2 = (d % DW) * 2;
      int gr = row0 + r, gk = k0 + c2;
      unsigned val = 0u;
      if (gr < rowMax && gk < kMax)      // K is even in every GEMM here
        val = *(const unsigned*)((const unsigned short*)src + (long long)gr * rs + gk);
      *(unsigned*)(dst + r * LDS_LD + c2) = val;
    }
  } else {                               // strided / f32 path, r fastest
    for (int idx = tid; idx < ROWS * KT; idx += 256) {
      int c = idx / ROWS, r = idx - c * ROWS;
      int gr = row0 + r, gk = k0 + c;
      unsigned short val = 0;
      if (gr < rowMax && gk < kMax) {
        long long o = (long long)gr * rs + (long long)gk * cs;
        val = srcF32 ? f2bf(((const float*)src)[o])
                     : ((const unsigned short*)src)[o];
      }
      dst[r * LDS_LD + c] = val;
    }
  }
}

// ---------------------------------------------------------------------------
// Generic batched GEMM: out[z][m,n] = act( sum_k A[z][m,k]*W[z][k,n] + bias )
// z = zb*Hc + zh decomposition supplies two independent offset strides.
// ---------------------------------------------------------------------------
struct GemmP {
  const void* A; long long a_r, a_c, sAb, sAh; int aF32;
  const void* W; long long w_k, w_n, sWb, sWh; int wF32;
  const float* bias;
  void* out; long long ldo, sOb, sOh;
  int M, N, K, Hc;
  int gcs;          // global column stride per zh (grouped FC: 97, else 0)
  long long gcm;    // global column bound (grouped FC: NC)
};

template<int RELU, int OBF>
__global__ __launch_bounds__(256) void gemm_bf16_kernel(GemmP p) {
  __shared__ unsigned short As[2][128 * LDS_LD];
  __shared__ unsigned short Bs[2][128 * LDS_LD];
  const int tid  = threadIdx.x;
  const int lane = tid & 31, wave = tid >> 5;
  const int waveM = wave >> 2, waveN = wave & 3;      // 2 x 4 wave grid
  const int z = blockIdx.z, zb = z / p.Hc, zh = z % p.Hc;
  const int bm = blockIdx.y * 128, bn = blockIdx.x * 128;

  const char* Ab = (const char*)p.A +
      ((long long)zb * p.sAb + (long long)zh * p.sAh) * (p.aF32 ? 4 : 2);
  const char* Wb = (const char*)p.W +
      ((long long)zb * p.sWb + (long long)zh * p.sWh) * (p.wF32 ? 4 : 2);
  const unsigned short* Aw = (const unsigned short*)Ab;
  const unsigned short* Ww = (const unsigned short*)Wb;

  // Async path legality: bf16, unit K-stride, 16B-aligned rows, full tile.
  const bool asyA = (!p.aF32) && (p.a_c == 1) && ((p.a_r & 7) == 0) &&
                    ((((size_t)Ab) & 15) == 0) && (bm + 128 <= p.M);
  const bool asyB = (!p.wF32) && (p.w_k == 1) && ((p.w_n & 7) == 0) &&
                    ((((size_t)Wb) & 15) == 0) && (bn + 128 <= p.N);

  v8f acc[4][2];
#pragma unroll
  for (int i = 0; i < 4; ++i)
#pragma unroll
    for (int j = 0; j < 2; ++j)
#pragma unroll
      for (int e = 0; e < 8; ++e) acc[i][j][e] = 0.0f;

  auto issue = [&](int kb, int buf) {
    if (asyA && kb + KT <= p.K)
      load_tile_async(As[buf], Aw + kb, p.a_r, bm, tid);
    else
      load_tile<128>(As[buf], Ab, p.aF32, p.a_r, p.a_c, bm, p.M, kb, p.K, tid);
    if (asyB && kb + KT <= p.K)
      load_tile_async(Bs[buf], Ww + kb, p.w_n, bn, tid);
    else
      load_tile<128>(Bs[buf], Wb, p.wF32, p.w_n, p.w_k, bn, p.N, kb, p.K, tid);
  };

  const int nk = (p.K + KT - 1) / KT;
  issue(0, 0);
  wait_async0();
  __syncthreads();

  for (int t = 0; t < nk; ++t) {
    const int cur = t & 1, nxt = cur ^ 1;
    if (t + 1 < nk) issue((t + 1) * KT, nxt);   // prefetch next tile (async)

    Frag af[4];
#pragma unroll
    for (int mt = 0; mt < 4; ++mt)
      af[mt] = load_afrag(As[cur] + (waveM * 64 + mt * 16) * LDS_LD, lane);
#pragma unroll
    for (int nt = 0; nt < 2; ++nt) {
      Frag bf = load_bfrag(Bs[cur] + (waveN * 32 + nt * 16) * LDS_LD, lane);
#pragma unroll
      for (int mt = 0; mt < 4; ++mt)
        acc[mt][nt] = __builtin_amdgcn_wmma_f32_16x16x32_bf16(
            false, af[mt].v, false, bf.v, (short)0, acc[mt][nt], false, false);
    }

    wait_async0();       // our prefetch landed in LDS
    __syncthreads();     // everyone done computing cur & filling nxt
  }

  // Epilogue: C layout — VGPR vv: rows vv (+8 for lanes 16-31), col = lane%16
  const int half = lane >> 4, nl = lane & 15;
  char* Ob = (char*)p.out +
      ((long long)zb * p.sOb + (long long)zh * p.sOh) * (OBF ? 2 : 4);
#pragma unroll
  for (int mt = 0; mt < 4; ++mt) {
#pragma unroll
    for (int nt = 0; nt < 2; ++nt) {
      int gn = bn + waveN * 32 + nt * 16 + nl;
      long long ngl = (long long)zh * p.gcs + gn;
      if (gn >= p.N || ngl >= p.gcm) continue;
      float bv = p.bias ? p.bias[ngl] : 0.0f;
      int gm0 = bm + waveM * 64 + mt * 16 + half * 8;
#pragma unroll
      for (int vv = 0; vv < 8; ++vv) {
        int gm = gm0 + vv;
        if (gm >= p.M) continue;
        float val = acc[mt][nt][vv] + bv;
        if (RELU) val = fmaxf(val, 0.0f);
        if (OBF) ((unsigned short*)Ob)[(long long)gm * p.ldo + gn] = f2bf(val);
        else     ((float*)Ob)[(long long)gm * p.ldo + gn] = val;
      }
    }
  }
}

// ---------------------------------------------------------------------------
// Small VALU kernels: conversions, LayerNorms, softmax
// ---------------------------------------------------------------------------
__global__ void cvt_kernel(const float* in, unsigned short* out, long long n) {
  long long i = (long long)blockIdx.x * blockDim.x + threadIdx.x;
  long long st = (long long)gridDim.x * blockDim.x;
  for (; i < n; i += st) out[i] = f2bf(in[i]);
}

DEV float blk_sum(float v, float* red) {
  int t = threadIdx.x;
  red[t] = v; __syncthreads();
#pragma unroll
  for (int s = 128; s > 0; s >>= 1) {
    if (t < s) red[t] += red[t + s];
    __syncthreads();
  }
  float r = red[0]; __syncthreads();
  return r;
}

// tgt = LN(2*query_embed) with (g1,be1); write f32 + bf16 copies. grid=G.
__global__ __launch_bounds__(256) void tgt_ln_kernel(
    const float* qe, const float* g1, const float* b1,
    float* tgtF, unsigned short* tgtB) {
  __shared__ float red[256];
  const int g = blockIdx.x, t = threadIdx.x;
  float y[3];
#pragma unroll
  for (int i = 0; i < 3; ++i) y[i] = 2.0f * qe[g * cfg::D + t + i * 256];
  float mean = blk_sum(y[0] + y[1] + y[2], red) * (1.0f / cfg::D);
  float sq = 0.f;
#pragma unroll
  for (int i = 0; i < 3; ++i) { float d = y[i] - mean; sq += d * d; }
  float inv = rsqrtf(blk_sum(sq, red) * (1.0f / cfg::D) + 1e-5f);
#pragma unroll
  for (int i = 0; i < 3; ++i) {
    int c = t + i * 256;
    float o = (y[i] - mean) * inv * g1[c] + b1[c];
    tgtF[g * cfg::D + c] = o;
    tgtB[g * cfg::D + c] = f2bf(o);
  }
}

// out = LN(a[row % aMod] + c[row]) * gamma + beta. grid = rows.
__global__ __launch_bounds__(256) void add_ln_kernel(
    const float* a, int aMod, const float* c,
    const float* gamma, const float* beta,
    float* outF, unsigned short* outB) {
  __shared__ float red[256];
  const int row = blockIdx.x, t = threadIdx.x;
  long long ra = (long long)(row % aMod) * cfg::D;
  long long rc = (long long)row * cfg::D;
  float y[3];
#pragma unroll
  for (int i = 0; i < 3; ++i) y[i] = a[ra + t + i * 256] + c[rc + t + i * 256];
  float mean = blk_sum(y[0] + y[1] + y[2], red) * (1.0f / cfg::D);
  float sq = 0.f;
#pragma unroll
  for (int i = 0; i < 3; ++i) { float d = y[i] - mean; sq += d * d; }
  float inv = rsqrtf(blk_sum(sq, red) * (1.0f / cfg::D) + 1e-5f);
#pragma unroll
  for (int i = 0; i < 3; ++i) {
    int cI = t + i * 256;
    float o = (y[i] - mean) * inv * gamma[cI] + beta[cI];
    if (outF) outF[rc + cI] = o;
    outB[rc + cI] = f2bf(o);
  }
}

// softmax over S=196 per (b,h,g); apply 1/sqrt(HD) here. grid = B*NH.
__global__ __launch_bounds__(256) void softmax_kernel(
    const float* scores, unsigned short* attn) {
  __shared__ float red[256];
  const int z = blockIdx.x, t = threadIdx.x;
  const float scale = 0.10206207261596577f;   // 1/sqrt(96)
  for (int g = 0; g < cfg::G; ++g) {
    long long base = ((long long)z * cfg::G + g) * cfg::S;
    float x = (t < cfg::S) ? scores[base + t] * scale : -3.0e38f;
    red[t] = x; __syncthreads();
#pragma unroll
    for (int s = 128; s > 0; s >>= 1) {
      if (t < s) red[t] = fmaxf(red[t], red[t + s]);
      __syncthreads();
    }
    float mx = red[0]; __syncthreads();
    float e = (t < cfg::S) ? __expf(x - mx) : 0.0f;
    float sm = blk_sum(e, red);
    if (t < cfg::S) attn[base + t] = f2bf(e / sm);
    __syncthreads();
  }
}

// ---------------------------------------------------------------------------
// Host side
// ---------------------------------------------------------------------------
static inline void run_gemm(const GemmP& p, int Z, bool relu, bool obf,
                            hipStream_t s) {
  dim3 grid((p.N + 127) / 128, (p.M + 127) / 128, Z);
  dim3 blk(256);
  if (relu) {
    if (obf) gemm_bf16_kernel<1, 1><<<grid, blk, 0, s>>>(p);
    else     gemm_bf16_kernel<1, 0><<<grid, blk, 0, s>>>(p);
  } else {
    if (obf) gemm_bf16_kernel<0, 1><<<grid, blk, 0, s>>>(p);
    else     gemm_bf16_kernel<0, 0><<<grid, blk, 0, s>>>(p);
  }
}

extern "C" void kernel_launch(void* const* d_in, const int* in_sizes, int n_in,
                              void* d_out, int out_size, void* d_ws,
                              size_t ws_size, hipStream_t stream) {
  using namespace cfg;
  (void)in_sizes; (void)n_in; (void)out_size; (void)ws_size;

  const float* x      = (const float*)d_in[0];
  const float* w_emb  = (const float*)d_in[1];
  const float* b_emb  = (const float*)d_in[2];
  const float* qemb   = (const float*)d_in[3];
  const float* wq = (const float*)d_in[4];  const float* bq = (const float*)d_in[5];
  const float* wk = (const float*)d_in[6];  const float* bk = (const float*)d_in[7];
  const float* wv = (const float*)d_in[8];  const float* bv = (const float*)d_in[9];
  const float* wo = (const float*)d_in[10]; const float* bo = (const float*)d_in[11];
  const float* g1 = (const float*)d_in[12]; const float* be1 = (const float*)d_in[13];
  const float* g2 = (const float*)d_in[14]; const float* be2 = (const float*)d_in[15];
  const float* g3 = (const float*)d_in[16]; const float* be3 = (const float*)d_in[17];
  const float* w1 = (const float*)d_in[18]; const float* bl1 = (const float*)d_in[19];
  const float* w2 = (const float*)d_in[20]; const float* bl2 = (const float*)d_in[21];
  const float* dup_pool = (const float*)d_in[22];
  const float* dup_bias = (const float*)d_in[23];
  float* out = (float*)d_out;

  // ---- workspace carve (all recomputed every call; ~262 MB) ----
  size_t off = 0;
  auto alloc = [&](size_t bytes) {
    size_t a = (off + 255) & ~(size_t)255;
    off = a + bytes;
    return (void*)((char*)d_ws + a);
  };
  typedef unsigned short u16;
  u16* wemb_b = (u16*)alloc((size_t)D * CIN * 2);
  u16* wq_b   = (u16*)alloc((size_t)D * D * 2);
  u16* wk_b   = (u16*)alloc((size_t)D * D * 2);
  u16* wv_b   = (u16*)alloc((size_t)D * D * 2);
  u16* wo_b   = (u16*)alloc((size_t)D * D * 2);
  u16* w1_b   = (u16*)alloc((size_t)FF * D * 2);
  u16* w2_b   = (u16*)alloc((size_t)D * FF * 2);
  u16* dup_b  = (u16*)alloc((size_t)G * D * DF * 2);
  float* tgtF = (float*)alloc((size_t)G * D * 4);
  u16* tgtB   = (u16*)alloc((size_t)G * D * 2);
  u16* q_b    = (u16*)alloc((size_t)G * D * 2);
  u16* mem_b  = (u16*)alloc((size_t)BS * D * 2);
  u16* k_b    = (u16*)alloc((size_t)BS * D * 2);
  u16* v_b    = (u16*)alloc((size_t)BS * D * 2);
  float* scF  = (float*)alloc((size_t)Bn * NH * G * S * 4);
  u16* attn_b = (u16*)alloc((size_t)Bn * NH * G * S * 2);
  u16* ctx_b  = (u16*)alloc((size_t)BG * D * 2);
  float* projF = (float*)alloc((size_t)BG * D * 4);
  float* tgt2F = (float*)alloc((size_t)BG * D * 4);
  u16* tgt2B   = (u16*)alloc((size_t)BG * D * 2);
  u16* ff_b    = (u16*)alloc((size_t)BG * FF * 2);
  float* ffoF  = (float*)alloc((size_t)BG * D * 4);
  u16* h_b     = (u16*)alloc((size_t)BG * D * 2);

  const long long BIG = 1ll << 60;

  // ---- 0) weight f32 -> bf16 (once per launch) ----
  auto cvt = [&](const float* src, u16* dst, long long n) {
    cvt_kernel<<<2048, 256, 0, stream>>>(src, dst, n);
  };
  cvt(w_emb, wemb_b, (long long)D * CIN);
  cvt(wq, wq_b, (long long)D * D);
  cvt(wk, wk_b, (long long)D * D);
  cvt(wv, wv_b, (long long)D * D);
  cvt(wo, wo_b, (long long)D * D);
  cvt(w1, w1_b, (long long)FF * D);
  cvt(w2, w2_b, (long long)D * FF);
  cvt(dup_pool, dup_b, (long long)G * D * DF);

  // ---- 1) tgt = LN(2*query_embed)  (batch-invariant) ----
  tgt_ln_kernel<<<G, 256, 0, stream>>>(qemb, g1, be1, tgtF, tgtB);

  // ---- 2) mem = relu(x^T @ w_embed^T + b_embed), read x[b,c,s] directly ----
  {
    GemmP p{};
    p.A = x; p.a_r = 1; p.a_c = S; p.sAb = (long long)CIN * S; p.sAh = 0; p.aF32 = 1;
    p.W = wemb_b; p.w_k = 1; p.w_n = CIN; p.sWb = 0; p.sWh = 0; p.wF32 = 0;
    p.bias = b_emb;
    p.out = mem_b; p.ldo = D; p.sOb = (long long)S * D; p.sOh = 0;
    p.M = S; p.N = D; p.K = CIN; p.Hc = 1; p.gcs = 0; p.gcm = BIG;
    run_gemm(p, Bn, /*relu*/true, /*bf16 out*/true, stream);
  }

  // ---- 3) k = mem@wk^T + bk ; v = mem@wv^T + bv ----
  {
    GemmP p{};
    p.A = mem_b; p.a_r = D; p.a_c = 1; p.sAb = 0; p.sAh = 0; p.aF32 = 0;
    p.W = wk_b; p.w_k = 1; p.w_n = D; p.sWb = 0; p.sWh = 0; p.wF32 = 0;
    p.bias = bk;
    p.out = k_b; p.ldo = D; p.sOb = 0; p.sOh = 0;
    p.M = BS; p.N = D; p.K = D; p.Hc = 1; p.gcs = 0; p.gcm = BIG;
    run_gemm(p, 1, false, true, stream);
    p.W = wv_b; p.bias = bv; p.out = v_b;
    run_gemm(p, 1, false, true, stream);
  }

  // ---- 4) q = tgt@wq^T + bq  (100 rows, once) ----
  {
    GemmP p{};
    p.A = tgtB; p.a_r = D; p.a_c = 1; p.sAb = 0; p.sAh = 0; p.aF32 = 0;
    p.W = wq_b; p.w_k = 1; p.w_n = D; p.sWb = 0; p.sWh = 0; p.wF32 = 0;
    p.bias = bq;
    p.out = q_b; p.ldo = D; p.sOb = 0; p.sOh = 0;
    p.M = G; p.N = D; p.K = D; p.Hc = 1; p.gcs = 0; p.gcm = BIG;
    run_gemm(p, 1, false, true, stream);
  }

  // ---- 5) scores[b,h] = q_h @ k_h^T  (512 batched 100x196x96 GEMMs) ----
  {
    GemmP p{};
    p.A = q_b; p.a_r = D; p.a_c = 1; p.sAb = 0; p.sAh = HD; p.aF32 = 0;
    p.W = k_b; p.w_k = 1; p.w_n = D;
    p.sWb = (long long)S * D; p.sWh = HD; p.wF32 = 0;
    p.bias = nullptr;
    p.out = scF; p.ldo = S;
    p.sOb = (long long)NH * G * S; p.sOh = (long long)G * S;
    p.M = G; p.N = S; p.K = HD; p.Hc = NH; p.gcs = 0; p.gcm = BIG;
    run_gemm(p, Bn * NH, false, /*f32 out*/false, stream);
  }

  // ---- 6) softmax (scale folded in) -> attn bf16 ----
  softmax_kernel<<<Bn * NH, 256, 0, stream>>>(scF, attn_b);

  // ---- 7) ctx[b,h] = attn @ v_h  (512 batched 100x96x196, K zero-padded) ----
  {
    GemmP p{};
    p.A = attn_b; p.a_r = S; p.a_c = 1;
    p.sAb = (long long)NH * G * S; p.sAh = (long long)G * S; p.aF32 = 0;
    p.W = v_b; p.w_k = D; p.w_n = 1;            // v in [K=s, N=hd] layout
    p.sWb = (long long)S * D; p.sWh = HD; p.wF32 = 0;
    p.bias = nullptr;
    p.out = ctx_b; p.ldo = D;
    p.sOb = (long long)G * D; p.sOh = HD;       // heads interleave into D
    p.M = G; p.N = HD; p.K = S; p.Hc = NH; p.gcs = 0; p.gcm = BIG;
    run_gemm(p, Bn * NH, false, true, stream);
  }

  // ---- 8) proj = ctx@wo^T + bo ; tgt2 = LN(tgt + proj) ----
  {
    GemmP p{};
    p.A = ctx_b; p.a_r = D; p.a_c = 1; p.sAb = 0; p.sAh = 0; p.aF32 = 0;
    p.W = wo_b; p.w_k = 1; p.w_n = D; p.sWb = 0; p.sWh = 0; p.wF32 = 0;
    p.bias = bo;
    p.out = projF; p.ldo = D; p.sOb = 0; p.sOh = 0;
    p.M = BG; p.N = D; p.K = D; p.Hc = 1; p.gcs = 0; p.gcm = BIG;
    run_gemm(p, 1, false, false, stream);
  }
  add_ln_kernel<<<BG, 256, 0, stream>>>(tgtF, G, projF, g2, be2, tgt2F, tgt2B);

  // ---- 9) FFN: ff = relu(tgt2@w1^T+bl1); ffo = ff@w2^T+bl2; h = LN(tgt2+ffo)
  {
    GemmP p{};
    p.A = tgt2B; p.a_r = D; p.a_c = 1; p.sAb = 0; p.sAh = 0; p.aF32 = 0;
    p.W = w1_b; p.w_k = 1; p.w_n = D; p.sWb = 0; p.sWh = 0; p.wF32 = 0;
    p.bias = bl1;
    p.out = ff_b; p.ldo = FF; p.sOb = 0; p.sOh = 0;
    p.M = BG; p.N = FF; p.K = D; p.Hc = 1; p.gcs = 0; p.gcm = BIG;
    run_gemm(p, 1, true, true, stream);
  }
  {
    GemmP p{};
    p.A = ff_b; p.a_r = FF; p.a_c = 1; p.sAb = 0; p.sAh = 0; p.aF32 = 0;
    p.W = w2_b; p.w_k = 1; p.w_n = FF; p.sWb = 0; p.sWh = 0; p.wF32 = 0;
    p.bias = bl2;
    p.out = ffoF; p.ldo = D; p.sOb = 0; p.sOh = 0;
    p.M = BG; p.N = D; p.K = FF; p.Hc = 1; p.gcs = 0; p.gcm = BIG;
    run_gemm(p, 1, false, false, stream);
  }
  add_ln_kernel<<<BG, 256, 0, stream>>>(tgt2F, 1 << 30, ffoF, g3, be3,
                                        (float*)nullptr, h_b);

  // ---- 10) grouped FC: logits[b, g*97+f] = h[b,g,:]·dup_pool[g,:,f] + bias
  {
    GemmP p{};
    p.A = h_b; p.a_r = (long long)G * D; p.a_c = 1;
    p.sAb = 0; p.sAh = D; p.aF32 = 0;
    p.W = dup_b; p.w_k = DF; p.w_n = 1;          // dup_pool[g] is [K=d, N=f]
    p.sWb = 0; p.sWh = (long long)D * DF; p.wF32 = 0;
    p.bias = dup_bias;                            // indexed by g*97+f
    p.out = out; p.ldo = NC; p.sOb = 0; p.sOh = DF;  // column shift per group
    p.M = Bn; p.N = DF; p.K = D; p.Hc = G;
    p.gcs = DF; p.gcm = NC;                      // clamp last group to NC
    run_gemm(p, G, false, false, stream);
  }
}
